// GATLayer_7292854469102
// MI455X (gfx1250) — matrix operand order repaired and checked by reading, not compiled, not run on hardware
//
#include <hip/hip_runtime.h>
#include <hip/hip_bf16.h>

// ---------------------------------------------------------------------------
// GAT layer for MI455X (gfx1250, wave32). Both GEMMs use
// v_wmma_f32_16x16x32_bf16 with 4-wide register blocking (4 WMMAs per
// A-tile), bf16 A/B, fp32 accumulate.
// ---------------------------------------------------------------------------

#define Bsz   8
#define Nn    1024
#define INF_  512
#define Hh    4
#define Dd    128
#define NEGI  (-1e9f)

typedef __attribute__((ext_vector_type(16))) __bf16 bf16x16;
typedef __attribute__((ext_vector_type(8)))  float  f32x8;

// A-matrix (16x32 bf16, MxK) per-lane K index for packed element t (0..15):
//   lanes 0-15 : v0-3 -> K 0..7,   v4-7 -> K 16..23
//   lanes 16-31: v0-3 -> K 8..15,  v4-7 -> K 24..31
__device__ __forceinline__ int a_kidx(int t, int half) {
    return (t & 7) + ((t >> 3) << 4) + (half << 3);
}
// B-matrix (32x16 bf16, KxN): same K order as A (lanes 0-15: K 0..7,16..23; lanes 16-31: K 8..15,24..31)
__device__ __forceinline__ int b_kidx(int t, int half) {
    return (t & 7) + ((t >> 3) << 4) + (half << 3);
}

// ---------------------------------------------------------------------------
// Kernel 1: h = x @ W   ([8192,512] x [512,512]).
// One wave per 16-row x 64-col output slab: A tile loaded once per K-step,
// reused by 4 WMMAs against 4 B column-tiles.
// ---------------------------------------------------------------------------
__global__ __launch_bounds__(256) void gemm_xw(const float* __restrict__ x,
                                               const float* __restrict__ W,
                                               float* __restrict__ hbuf) {
    const int lane = threadIdx.x & 31;
    const int wave = threadIdx.x >> 5;
    const int tile = blockIdx.x * 8 + wave;      // 512 row-tiles * 8 col-groups = 4096
    const int rt = tile >> 3;                    // 0..511
    const int cg = tile & 7;                     // 0..7 (64-col group)
    const int row0 = rt * 16;
    const int col0 = cg * 64;
    const int ln = lane & 15, half = lane >> 4;

    const float* __restrict__ xrow = x + (size_t)(row0 + ln) * INF_;

    f32x8 acc0 = {}, acc1 = {}, acc2 = {}, acc3 = {};
    for (int k0 = 0; k0 < INF_; k0 += 32) {
        if (k0 + 32 < INF_) {
            __builtin_prefetch(xrow + k0 + 32, 0, 0);               // global_prefetch_b8
            __builtin_prefetch(W + (size_t)(k0 + 32) * INF_ + col0, 0, 0);
        }
        bf16x16 a, b0, b1, b2, b3;
#pragma unroll
        for (int t = 0; t < 16; ++t) {
            a[t] = (__bf16)xrow[k0 + a_kidx(t, half)];
            const float* __restrict__ wr =
                W + (size_t)(k0 + b_kidx(t, half)) * INF_ + col0 + ln;
            b0[t] = (__bf16)wr[0];
            b1[t] = (__bf16)wr[16];
            b2[t] = (__bf16)wr[32];
            b3[t] = (__bf16)wr[48];
        }
        acc0 = __builtin_amdgcn_wmma_f32_16x16x32_bf16(false, a, false, b0, (short)0, acc0, false, false);
        acc1 = __builtin_amdgcn_wmma_f32_16x16x32_bf16(false, a, false, b1, (short)0, acc1, false, false);
        acc2 = __builtin_amdgcn_wmma_f32_16x16x32_bf16(false, a, false, b2, (short)0, acc2, false, false);
        acc3 = __builtin_amdgcn_wmma_f32_16x16x32_bf16(false, a, false, b3, (short)0, acc3, false, false);
    }
#pragma unroll
    for (int r = 0; r < 8; ++r) {
        const int row = row0 + r + half * 8;     // C/D layout: M = r + half*8
        float* __restrict__ orow = hbuf + (size_t)row * INF_ + col0 + ln;
        orow[0]  = acc0[r];
        orow[16] = acc1[r];
        orow[32] = acc2[r];
        orow[48] = acc3[r];
    }
}

// ---------------------------------------------------------------------------
// Kernel 2: e_i / e_j head-dot-products. One wave per (b,n,h).
// ---------------------------------------------------------------------------
__global__ __launch_bounds__(256) void calc_e(const float* __restrict__ hbuf,
                                              const float* __restrict__ a_src,
                                              const float* __restrict__ a_dst,
                                              float* __restrict__ ei,
                                              float* __restrict__ ej) {
    const int gw   = (blockIdx.x * blockDim.x + threadIdx.x) >> 5; // 32768 waves
    const int lane = threadIdx.x & 31;
    const int h  = gw & (Hh - 1);
    const int bn = gw >> 2;
    const float* __restrict__ hrow = hbuf + (size_t)bn * INF_ + h * Dd;
    float ss = 0.f, sd = 0.f;
    for (int d = lane; d < Dd; d += 32) {
        float v = hrow[d];
        ss += v * a_src[h * Dd + d];
        sd += v * a_dst[h * Dd + d];
    }
#pragma unroll
    for (int o = 16; o; o >>= 1) {
        ss += __shfl_xor(ss, o, 32);
        sd += __shfl_xor(sd, o, 32);
    }
    if (lane == 0) { ei[gw] = ss; ej[gw] = sd; }
}

__device__ __forceinline__ float logit_val(int maskI, int maskJ, float eiv, float ejv) {
    if (maskI && maskJ) {
        float z = eiv + ejv;
        return z > 0.f ? z : 0.2f * z;           // leaky_relu(0.2)
    }
    return NEGI;                                 // finite, so fully-masked rows -> uniform
}

// ---------------------------------------------------------------------------
// Kernel 3: per-(b,i,h) softmax row max and 1/sum. One wave per row.
// ---------------------------------------------------------------------------
__global__ __launch_bounds__(256) void softmax_stats(const int* __restrict__ mask,
                                                     const float* __restrict__ ei,
                                                     const float* __restrict__ ej,
                                                     float* __restrict__ mrow,
                                                     float* __restrict__ rsrow) {
    const int gw   = (blockIdx.x * blockDim.x + threadIdx.x) >> 5; // 32768 waves
    const int lane = threadIdx.x & 31;
    const int h  = gw & (Hh - 1);
    const int bn = gw >> 2;
    const int b  = bn >> 10;
    const int i  = bn & (Nn - 1);
    const int maskI = mask[b * Nn + i];
    const float eiv = ei[gw];

    float mx = NEGI;
    for (int j = lane; j < Nn; j += 32)
        mx = fmaxf(mx, logit_val(maskI, mask[b * Nn + j], eiv, ej[(b * Nn + j) * Hh + h]));
#pragma unroll
    for (int o = 16; o; o >>= 1) mx = fmaxf(mx, __shfl_xor(mx, o, 32));

    float s = 0.f;
    for (int j = lane; j < Nn; j += 32)
        s += __expf(logit_val(maskI, mask[b * Nn + j], eiv, ej[(b * Nn + j) * Hh + h]) - mx);
#pragma unroll
    for (int o = 16; o; o >>= 1) s += __shfl_xor(s, o, 32);

    if (lane == 0) { mrow[gw] = mx; rsrow[gw] = 1.f / s; }
}

// ---------------------------------------------------------------------------
// Kernel 4: h' = P @ h per (b,head). Probability A-tiles regenerated in
// registers (flash-attention style, rank-1 logits) and amortized over 4
// d-tiles = 4 WMMAs per A-tile. Block = 2 waves covering all of D=128.
// Writes h' into d_out.
// ---------------------------------------------------------------------------
__global__ __launch_bounds__(64) void attn_gemm(const float* __restrict__ hbuf,
                                                const int* __restrict__ mask,
                                                const float* __restrict__ ei,
                                                const float* __restrict__ ej,
                                                const float* __restrict__ mrow,
                                                const float* __restrict__ rsrow,
                                                float* __restrict__ out) {
    const int blk  = blockIdx.x;                 // B*H*64 = 2048
    const int it   = blk & 63;                   // i-tile within (b,head)
    const int bh   = blk >> 6;
    const int head = bh & (Hh - 1);
    const int b    = bh >> 2;
    const int wave = threadIdx.x >> 5;           // 0..1
    const int lane = threadIdx.x & 31;
    const int d0   = wave * 64;                  // each wave: 4 d-tiles of 16
    const int ln   = lane & 15, half = lane >> 4;

    const int   i     = it * 16 + ln;            // A row owned by this lane
    const int   gi    = (b * Nn + i) * Hh + head;
    const int   maskI = mask[b * Nn + i];
    const float eiv = ei[gi], mrv = mrow[gi], rsv = rsrow[gi];

    const float* __restrict__ hb = hbuf + head * Dd + d0 + ln;

    f32x8 acc0 = {}, acc1 = {}, acc2 = {}, acc3 = {};
    for (int j0 = 0; j0 < Nn; j0 += 32) {
        if (j0 + 32 < Nn)
            __builtin_prefetch(hb + (size_t)(b * Nn + j0 + 32) * INF_, 0, 0);
        bf16x16 a, b0, b1, b2, b3;
#pragma unroll
        for (int t = 0; t < 16; ++t) {
            const int ja = j0 + a_kidx(t, half);
            const float v = logit_val(maskI, mask[b * Nn + ja], eiv,
                                      ej[(b * Nn + ja) * Hh + head]);
            a[t] = (__bf16)(__expf(v - mrv) * rsv);

            const int jb = j0 + b_kidx(t, half);
            const float* __restrict__ hr = hb + (size_t)(b * Nn + jb) * INF_;
            b0[t] = (__bf16)hr[0];
            b1[t] = (__bf16)hr[16];
            b2[t] = (__bf16)hr[32];
            b3[t] = (__bf16)hr[48];
        }
        acc0 = __builtin_amdgcn_wmma_f32_16x16x32_bf16(false, a, false, b0, (short)0, acc0, false, false);
        acc1 = __builtin_amdgcn_wmma_f32_16x16x32_bf16(false, a, false, b1, (short)0, acc1, false, false);
        acc2 = __builtin_amdgcn_wmma_f32_16x16x32_bf16(false, a, false, b2, (short)0, acc2, false, false);
        acc3 = __builtin_amdgcn_wmma_f32_16x16x32_bf16(false, a, false, b3, (short)0, acc3, false, false);
    }
#pragma unroll
    for (int r = 0; r < 8; ++r) {
        const int row = it * 16 + r + half * 8;
        float* __restrict__ orow =
            out + (size_t)(b * Nn + row) * INF_ + head * Dd + d0 + ln;
        orow[0]  = acc0[r];
        orow[16] = acc1[r];
        orow[32] = acc2[r];
        orow[48] = acc3[r];
    }
}

// ---------------------------------------------------------------------------
// Kernel 5: out = LayerNorm(out + x) * gamma + beta, in place, one block/row.
// ---------------------------------------------------------------------------
__global__ __launch_bounds__(256) void residual_ln(const float* __restrict__ x,
                                                   const float* __restrict__ gamma,
                                                   const float* __restrict__ beta,
                                                   float* __restrict__ out) {
    const int row = blockIdx.x;                  // 8192 rows
    const int tid = threadIdx.x;
    __shared__ float red[256];

    const size_t base = (size_t)row * INF_;
    float y0 = out[base + tid]       + x[base + tid];
    float y1 = out[base + tid + 256] + x[base + tid + 256];

    red[tid] = y0 + y1;
    __syncthreads();
    for (int o = 128; o; o >>= 1) { if (tid < o) red[tid] += red[tid + o]; __syncthreads(); }
    const float mu = red[0] * (1.0f / INF_);
    __syncthreads();

    const float d0 = y0 - mu, d1 = y1 - mu;
    red[tid] = d0 * d0 + d1 * d1;
    __syncthreads();
    for (int o = 128; o; o >>= 1) { if (tid < o) red[tid] += red[tid + o]; __syncthreads(); }
    const float rstd = rsqrtf(red[0] * (1.0f / INF_) + 1e-5f);

    out[base + tid]       = d0 * rstd * gamma[tid]       + beta[tid];
    out[base + tid + 256] = d1 * rstd * gamma[tid + 256] + beta[tid + 256];
}

// ---------------------------------------------------------------------------
extern "C" void kernel_launch(void* const* d_in, const int* in_sizes, int n_in,
                              void* d_out, int out_size, void* d_ws, size_t ws_size,
                              hipStream_t stream) {
    (void)in_sizes; (void)n_in; (void)out_size; (void)ws_size;
    const float* x     = (const float*)d_in[0];
    const int*   mask  = (const int*)  d_in[1];
    const float* W     = (const float*)d_in[2];
    const float* a_src = (const float*)d_in[3];
    const float* a_dst = (const float*)d_in[4];
    const float* gamma = (const float*)d_in[5];
    const float* beta  = (const float*)d_in[6];
    float* out = (float*)d_out;

    char*  ws    = (char*)d_ws;
    float* hbuf  = (float*)ws;                                   // 8192*512 f32 = 16 MiB
    float* ei    = (float*)(ws + (size_t)Bsz * Nn * INF_ * 4);   // 32768 f32
    float* ej    = ei + Bsz * Nn * Hh;
    float* mrow  = ej + Bsz * Nn * Hh;
    float* rsrow = mrow + Bsz * Nn * Hh;

    gemm_xw      <<<512, 256, 0, stream>>>(x, W, hbuf);          // 4096 waves
    calc_e       <<<4096, 256, 0, stream>>>(hbuf, a_src, a_dst, ei, ej);
    softmax_stats<<<4096, 256, 0, stream>>>(mask, ei, ej, mrow, rsrow);
    attn_gemm    <<<2048, 64, 0, stream>>>(hbuf, mask, ei, ej, mrow, rsrow, out);
    residual_ln  <<<8192, 256, 0, stream>>>(x, gamma, beta, out);
}
